// MultiHeadAttentionLayer_67035849556294
// MI455X (gfx1250) — matrix operand order repaired and checked by reading, not compiled, not run on hardware
//
#include <hip/hip_runtime.h>
#include <hip/hip_bf16.h>

typedef unsigned short h16;
typedef unsigned int u32;
typedef __attribute__((ext_vector_type(16))) __bf16 v16bf;
typedef __attribute__((ext_vector_type(8))) float v8f;

union FragB { v16bf v; h16 h[16]; u32 w[8]; };
union AccF  { v8f  v; float f[8]; };

__device__ __forceinline__ h16 f2bf(float x) {
  u32 u = __float_as_uint(x);
  u = u + 0x7FFFu + ((u >> 16) & 1u);   // round-to-nearest-even
  return (h16)(u >> 16);
}
__device__ __forceinline__ float bf2f(h16 h) {
  return __uint_as_float(((u32)h) << 16);
}

// ---- WMMA fragment loaders (ISA 7.12.2, 16-bit, wave32) --------------------
// A (16x32, MxK), row-major source with row stride ld (elements).
// lane l: M = l&15; half=l>>4.  VGPR r holds K pair:
//   K = 2r + 16*half
__device__ __forceinline__ v16bf load_a16(const h16* base, int ld, int lane) {
  FragB f;
  const int m = lane & 15, hf = lane >> 4;
  const h16* row = base + (size_t)m * ld;
#pragma unroll
  for (int r = 0; r < 8; ++r) {
    const int kb = 2 * r + 16 * hf;
    f.w[r] = *(const u32*)(row + kb);
  }
  return f.v;
}

// B (32x16, KxN) where source is row-major [N][K] (i.e. B[k][n] = src[n*ld+k]);
// lanes 0-15: K=0..15 (VGPR r -> K=2r,2r+1); lanes 16-31: K=16..31.
__device__ __forceinline__ v16bf load_b16_kc(const h16* base, int ld, int lane) {
  FragB f;
  const int n = lane & 15, hf = lane >> 4;
  const h16* p = base + (size_t)n * ld + 16 * hf;
#pragma unroll
  for (int r = 0; r < 8; ++r) f.w[r] = *(const u32*)(p + 2 * r);
  return f.v;
}

// Output store modes for the projection GEMM.
enum StoreMode { OUT_F32 = 0, OUT_HEAD_BF16 = 1, OUT_HEAD_T_BF16 = 2 };

// ---- GEMM: D = A @ W^T + bias.  A:[M,K], W:[N,K] fp32.  64x64 block tile,
// 4 waves x (16 x 64) per wave, K-step 32, bf16 LDS staging, fp32 accum. -----
template <bool A_F32, int SMODE>
__global__ __launch_bounds__(128) void gemm_xwt(
    const void* __restrict__ Ap, const float* __restrict__ W,
    const float* __restrict__ bias, void* __restrict__ Dst,
    int M, int N, int K) {
  __shared__ h16 As[64 * 32];
  __shared__ h16 Bs[64 * 32];
  const int tid = threadIdx.x;
  const int lane = tid & 31, wave = tid >> 5;
  const int n0 = blockIdx.x * 64, m0 = blockIdx.y * 64;

  AccF acc[4];
#pragma unroll
  for (int t = 0; t < 4; ++t)
#pragma unroll
    for (int j = 0; j < 8; ++j) acc[t].f[j] = 0.f;

  for (int k0 = 0; k0 < K; k0 += 32) {
    __syncthreads();
    for (int i = tid; i < 64 * 32; i += 128) {
      const int r = i >> 5, c = i & 31;
      if constexpr (A_F32)
        As[i] = f2bf(((const float*)Ap)[(size_t)(m0 + r) * K + k0 + c]);
      else
        As[i] = ((const h16*)Ap)[(size_t)(m0 + r) * K + k0 + c];
      Bs[i] = f2bf(W[(size_t)(n0 + r) * K + k0 + c]);
    }
    __syncthreads();
    const v16bf a = load_a16(As + wave * 16 * 32, 32, lane);
#pragma unroll
    for (int t = 0; t < 4; ++t) {
      const v16bf b = load_b16_kc(Bs + t * 16 * 32, 32, lane);
      acc[t].v = __builtin_amdgcn_wmma_f32_16x16x32_bf16(
          false, a, false, b, (short)0, acc[t].v, false, false);
    }
  }

  const int hf = lane >> 4, nl = lane & 15;
#pragma unroll
  for (int t = 0; t < 4; ++t) {
#pragma unroll
    for (int j = 0; j < 8; ++j) {
      const int gm = m0 + wave * 16 + j + 8 * hf;
      const int gn = n0 + t * 16 + nl;
      const float v = acc[t].f[j] + bias[gn];
      if constexpr (SMODE == OUT_HEAD_BF16) {
        // [B*L, HID] -> head-major bf16 [B*H, L, 64]
        const int b = gm >> 10, q = gm & 1023, h = gn >> 6, d = gn & 63;
        ((h16*)Dst)[(((size_t)(b * 16 + h)) * 1024 + q) * 64 + d] = f2bf(v);
      } else if constexpr (SMODE == OUT_HEAD_T_BF16) {
        // [B*L, HID] -> transposed head-major bf16 [B*H, 64, L]  (for V)
        const int b = gm >> 10, q = gm & 1023, h = gn >> 6, d = gn & 63;
        ((h16*)Dst)[(((size_t)(b * 16 + h)) * 64 + d) * 1024 + q] = f2bf(v);
      } else {
        ((float*)Dst)[(size_t)gm * N + gn] = v;
      }
    }
  }
}

// ---- scores = sigmoid(gat)*prev + (1-sigmoid(gat)) * (Q K^T) ---------------
__global__ __launch_bounds__(128) void scores_gate(
    const h16* __restrict__ Q, const h16* __restrict__ Kh,
    const float* __restrict__ prev, const float* __restrict__ gat,
    float* __restrict__ scores) {
  const int lane = threadIdx.x & 31, wave = threadIdx.x >> 5;
  const int bh = blockIdx.z;
  const int n0 = blockIdx.x * 64;
  const int m0 = blockIdx.y * 64 + wave * 16;
  const h16* qb = Q + ((size_t)bh * 1024 + m0) * 64;
  const h16* kb = Kh + ((size_t)bh * 1024 + n0) * 64;

  AccF acc[4];
#pragma unroll
  for (int t = 0; t < 4; ++t)
#pragma unroll
    for (int j = 0; j < 8; ++j) acc[t].f[j] = 0.f;

  const v16bf a0 = load_a16(qb, 64, lane);
  const v16bf a1 = load_a16(qb + 32, 64, lane);
#pragma unroll
  for (int t = 0; t < 4; ++t) {
    const v16bf b0 = load_b16_kc(kb + (size_t)t * 16 * 64, 64, lane);
    const v16bf b1 = load_b16_kc(kb + (size_t)t * 16 * 64 + 32, 64, lane);
    acc[t].v = __builtin_amdgcn_wmma_f32_16x16x32_bf16(
        false, a0, false, b0, (short)0, acc[t].v, false, false);
    acc[t].v = __builtin_amdgcn_wmma_f32_16x16x32_bf16(
        false, a1, false, b1, (short)0, acc[t].v, false, false);
  }

  const float g = 1.f / (1.f + __expf(-gat[0]));
  const int hf = lane >> 4, nl = lane & 15;
#pragma unroll
  for (int t = 0; t < 4; ++t) {
#pragma unroll
    for (int j = 0; j < 8; ++j) {
      const int gm = m0 + j + 8 * hf;
      const int gn = n0 + t * 16 + nl;
      const size_t idx = ((size_t)bh * 1024 + gm) * 1024 + gn;
      const float pv = __builtin_nontemporal_load(prev + idx);  // streamed once
      scores[idx] = pv * g + (1.f - g) * acc[t].f[j];
    }
  }
}

// ---- fused mask + softmax + P@V: 1 wave handles 16 query rows. -------------
// V is pre-transposed per head: Vt[BH][64][1024], so B-fragments are
// k-contiguous (vectorizes to b128 loads).
__global__ __launch_bounds__(32) void softmax_pv(
    const float* __restrict__ scores, const h16* __restrict__ Vt,
    const int* __restrict__ mask, h16* __restrict__ att) {
  __shared__ h16 sp[16 * 1024];      // masked scores, then unnormalized probs
  __shared__ float rmax[16], rsum[16];
  const int lane = threadIdx.x;
  const int q0 = blockIdx.x * 16;
  const int bh = blockIdx.y;
  const int b = bh >> 4, h = bh & 15;

  // phase 1: load + mask + scale (1/sqrt(64)); per-row max
  for (int r = 0; r < 16; ++r) {
    const int q = q0 + r;
    const int len = mask[(bh & 3) * 1024 + q];  // reference tiles mask: row = bh % B
    const float* srow = scores + ((size_t)bh * 1024 + q) * 1024;
    float mx = -3.0e38f;
    for (int c = lane; c < 1024; c += 32) {
      const float s = (c < len) ? srow[c] * 0.125f : -1.25e8f;  // -1e9/8
      mx = fmaxf(mx, s);
      sp[r * 1024 + c] = f2bf(s);
    }
#pragma unroll
    for (int off = 16; off > 0; off >>= 1) mx = fmaxf(mx, __shfl_xor(mx, off, 32));
    if (lane == 0) rmax[r] = mx;
  }
  __syncthreads();

  // phase 2: exp(s - max) in-place (bf16), per-row sum
  for (int r = 0; r < 16; ++r) {
    const float mx = rmax[r];
    float sum = 0.f;
    for (int c = lane; c < 1024; c += 32) {
      const float p = __expf(bf2f(sp[r * 1024 + c]) - mx);
      sum += p;
      sp[r * 1024 + c] = f2bf(p);
    }
#pragma unroll
    for (int off = 16; off > 0; off >>= 1) sum += __shfl_xor(sum, off, 32);
    if (lane == 0) rsum[r] = sum;
  }
  __syncthreads();

  // phase 3: att[16,64] = P_unnorm[16,1024] @ V[1024,64]; 1/sum in epilogue
  AccF acc[4];
#pragma unroll
  for (int t = 0; t < 4; ++t)
#pragma unroll
    for (int j = 0; j < 8; ++j) acc[t].f[j] = 0.f;

  const h16* vb = Vt + (size_t)bh * 64 * 1024;   // [64 d-rows][1024 k]
  for (int k0 = 0; k0 < 1024; k0 += 32) {
    const v16bf a = load_a16(sp + k0, 1024, lane);
#pragma unroll
    for (int t = 0; t < 4; ++t) {
      // B[k][n] = Vt[(t*16+n)*1024 + k0 + k]  -> k-contiguous loader
      const v16bf bf = load_b16_kc(vb + (size_t)(t * 16) * 1024 + k0, 1024, lane);
      acc[t].v = __builtin_amdgcn_wmma_f32_16x16x32_bf16(
          false, a, false, bf, (short)0, acc[t].v, false, false);
    }
  }

  const int hf = lane >> 4, nl = lane & 15;
#pragma unroll
  for (int t = 0; t < 4; ++t) {
#pragma unroll
    for (int j = 0; j < 8; ++j) {
      const int row = j + 8 * hf;
      const int q = q0 + row;
      const int d = t * 16 + nl;
      const float v = acc[t].f[j] / rsum[row];
      // merged heads: [B*L, HID] bf16, col = h*64 + d
      att[((size_t)(b * 1024 + q)) * 1024 + h * 64 + d] = f2bf(v);
    }
  }
}

extern "C" void kernel_launch(void* const* d_in, const int* in_sizes, int n_in,
                              void* d_out, int out_size, void* d_ws, size_t ws_size,
                              hipStream_t stream) {
  const float* queries = (const float*)d_in[0];
  const float* keys    = (const float*)d_in[1];
  const float* values  = (const float*)d_in[2];
  const float* prev    = (const float*)d_in[3];
  const int*   mask    = (const int*)d_in[4];
  const float* Wq_w = (const float*)d_in[5];
  const float* Wq_b = (const float*)d_in[6];
  const float* Wk_w = (const float*)d_in[7];
  const float* Wk_b = (const float*)d_in[8];
  const float* Wv_w = (const float*)d_in[9];
  const float* Wv_b = (const float*)d_in[10];
  const float* Wo_w = (const float*)d_in[11];
  const float* Wo_b = (const float*)d_in[12];
  const float* gat  = (const float*)d_in[13];

  float* out    = (float*)d_out;                        // [4,1024,1024]
  float* scores = out + (size_t)4 * 1024 * 1024;        // [64,1024,1024]

  // workspace: Q,K head-major bf16, V transposed head-major bf16, att bf16
  h16* qh  = (h16*)d_ws;
  h16* kh  = qh + (size_t)64 * 1024 * 64;
  h16* vt  = kh + (size_t)64 * 1024 * 64;
  h16* att = vt + (size_t)64 * 1024 * 64;

  const dim3 gproj(16, 64);  // N/64, M/64  (M=4096, N=K=1024)
  gemm_xwt<true, OUT_HEAD_BF16><<<gproj, 128, 0, stream>>>(
      queries, Wq_w, Wq_b, qh, 4096, 1024, 1024);
  gemm_xwt<true, OUT_HEAD_BF16><<<gproj, 128, 0, stream>>>(
      keys, Wk_w, Wk_b, kh, 4096, 1024, 1024);
  gemm_xwt<true, OUT_HEAD_T_BF16><<<gproj, 128, 0, stream>>>(
      values, Wv_w, Wv_b, vt, 4096, 1024, 1024);

  const dim3 gs(16, 16, 64);  // n-tiles, m-tiles, BH
  scores_gate<<<gs, 128, 0, stream>>>(qh, kh, prev, gat, scores);

  const dim3 gsm(64, 64);     // 16-row blocks, BH
  softmax_pv<<<gsm, 32, 0, stream>>>(scores, vt, mask, att);

  gemm_xwt<false, OUT_F32><<<gproj, 128, 0, stream>>>(
      att, Wo_w, Wo_b, out, 4096, 1024, 1024);
}
